// TrentoDGN_27676769255860
// MI455X (gfx1250) — matrix operand order repaired and checked by reading, not compiled, not run on hardware
//
#include <hip/hip_runtime.h>
#include <hip/hip_bf16.h>
#include <stdint.h>

typedef __attribute__((ext_vector_type(16))) __bf16 v16bf;
typedef __attribute__((ext_vector_type(8)))  __bf16 v8bf;
typedef __attribute__((ext_vector_type(4)))  __bf16 v4bf;
typedef __attribute__((ext_vector_type(8)))  float  v8f;

#define TPB 256

__global__ void k_zero(float* __restrict__ p, long cnt) {
  long t = (long)blockIdx.x * blockDim.x + threadIdx.x;
  if (t < cnt) p[t] = 0.0f;
}

// Pack Wl (layer, k, col) fp32 -> per-layer transposed bf16: PW[layer][col][k].
// Each lane's WMMA B operand is then 16 consecutive K (the contiguous k order this file uses).
__global__ void k_packw(const float* __restrict__ Wl, __bf16* __restrict__ out, int cnt) {
  int t = blockIdx.x * blockDim.x + threadIdx.x;
  if (t >= cnt) return;
  int layer = t >> 12;          // /4096
  int r = t & 4095;
  int col = r >> 6;
  int k = r & 63;
  out[t] = (__bf16)Wl[(layer << 12) + k * 64 + col];
}

__global__ void k_degree(const int* __restrict__ dst, float* __restrict__ cnt, int e) {
  int t = blockIdx.x * blockDim.x + threadIdx.x;
  if (t < e) atomicAdd(&cnt[dst[t]], 1.0f);
}

__global__ void k_invdeg(float* __restrict__ c, int n) {
  int t = blockIdx.x * blockDim.x + threadIdx.x;
  if (t < n) c[t] = 1.0f / fmaxf(c[t], 1.0f);
}

// z[n,c] = sum_k x10[n,k]*W0[k,c] + b0[c];  x10 = [x[:, :9], sites]
__global__ void k_input(const float* __restrict__ x, const int* __restrict__ sites,
                        const float* __restrict__ W0, const float* __restrict__ b0,
                        float* __restrict__ z, float* __restrict__ sum,
                        float* __restrict__ sumsq, int n) {
  long t = (long)blockIdx.x * blockDim.x + threadIdx.x;
  if (t >= (long)n * 64) return;
  int i = (int)(t >> 6), c = (int)(t & 63);
  const float* xr = x + (size_t)i * 10;
  float acc = b0[c];
#pragma unroll
  for (int k = 0; k < 9; ++k) acc = fmaf(xr[k], W0[k * 64 + c], acc);
  acc = fmaf((float)sites[i], W0[9 * 64 + c], acc);
  z[t] = acc;
  atomicAdd(&sum[c], acc);
  atomicAdd(&sumsq[c], acc * acc);
}

__global__ void k_finalize(const float* __restrict__ sum, const float* __restrict__ sumsq,
                           const float* __restrict__ g, const float* __restrict__ beta,
                           float* __restrict__ scale, float* __restrict__ shift, int n) {
  int c = threadIdx.x;  // 64 threads
  float invn = 1.0f / (float)n;
  float mu = sum[c] * invn;
  float var = sumsq[c] * invn - mu * mu;
  float sc = g[c] * rsqrtf(var + 1e-5f);
  scale[c] = sc;
  shift[c] = beta[c] - mu * sc;
}

// h = relu(z*scale + shift); partial[n] += dot(h_row, Wp_segment)
__global__ void k_bnrelu(const float* __restrict__ Z, const float* __restrict__ scale,
                         const float* __restrict__ shift, float* __restrict__ H,
                         float* __restrict__ partial, const float* __restrict__ wp, int n) {
  int i = blockIdx.x * blockDim.x + threadIdx.x;
  if (i >= n) return;
  const float4* zr = (const float4*)(Z + (size_t)i * 64);
  float4* hr = (float4*)(H + (size_t)i * 64);
  const float4* sc4 = (const float4*)scale;
  const float4* sh4 = (const float4*)shift;
  const float4* wp4 = (const float4*)wp;
  float dot = 0.0f;
#pragma unroll
  for (int q = 0; q < 16; ++q) {
    float4 z4 = zr[q], sc = sc4[q], sh = sh4[q], w4 = wp4[q], h4;
    h4.x = fmaxf(0.0f, fmaf(z4.x, sc.x, sh.x));
    h4.y = fmaxf(0.0f, fmaf(z4.y, sc.y, sh.y));
    h4.z = fmaxf(0.0f, fmaf(z4.z, sc.z, sh.z));
    h4.w = fmaxf(0.0f, fmaf(z4.w, sc.w, sh.w));
    dot = fmaf(h4.x, w4.x, dot);
    dot = fmaf(h4.y, w4.y, dot);
    dot = fmaf(h4.z, w4.z, dot);
    dot = fmaf(h4.w, w4.w, dot);
    hr[q] = h4;
  }
  partial[i] += dot;
}

// agg[dst] += h[src] * edge_attr[e,0] ; 16 threads/edge, 4 feats each
__global__ void k_scatter(const int* __restrict__ src, const int* __restrict__ dst,
                          const float* __restrict__ ea, const float* __restrict__ H,
                          float* __restrict__ agg, int e) {
  long t = (long)blockIdx.x * blockDim.x + threadIdx.x;
  if (t >= (long)e * 16) return;
  int ed = (int)(t >> 4);
  int c = (int)(t & 15) * 4;
  int s = src[ed], d = dst[ed];
  float w = ea[(size_t)ed * 4];
  float4 hv = *(const float4*)(H + (size_t)s * 64 + c);
  float* ap = agg + (size_t)d * 64 + c;
  atomicAdd(ap + 0, hv.x * w);
  atomicAdd(ap + 1, hv.y * w);
  atomicAdd(ap + 2, hv.z * w);
  atomicAdd(ap + 3, hv.w * w);
}

// T(bf16) = (1+eps)*H + agg * inv_degree   -- conversion fused here, off the gemm hot path
__global__ void k_combine(const float* __restrict__ H, const float* __restrict__ agg,
                          const float* __restrict__ inv, const float* __restrict__ eps_p,
                          __bf16* __restrict__ T, int n) {
  long t = (long)blockIdx.x * blockDim.x + threadIdx.x;
  if (t >= (long)n * 16) return;
  int i = (int)(t >> 4);
  float e1 = 1.0f + *eps_p;
  float iv = inv[i];
  float4 h4 = ((const float4*)H)[t];
  float4 a4 = ((const float4*)agg)[t];
  v4bf o;
  o[0] = (__bf16)fmaf(e1, h4.x, a4.x * iv);
  o[1] = (__bf16)fmaf(e1, h4.y, a4.y * iv);
  o[2] = (__bf16)fmaf(e1, h4.z, a4.z * iv);
  o[3] = (__bf16)fmaf(e1, h4.w, a4.w * iv);
  ((v4bf*)T)[t] = o;
}

// Z[N,64] = T[N,64](bf16) @ W[64,64] + bias ; v_wmma_f32_16x16x32_bf16, f32 accumulate.
// Block: 8 waves = 2 (M) x 4 (N) tiles of 16x16; K=64 -> two WMMAs.
// Wb is the transposed bf16 weight (PW[col][k]); each lane's B operand is 32B contiguous.
__global__ __launch_bounds__(256)
void k_wmma_gemm(const __bf16* __restrict__ T, const __bf16* __restrict__ Wb,
                 const float* __restrict__ bias, float* __restrict__ Z,
                 float* __restrict__ sum, float* __restrict__ sumsq, int n) {
  __shared__ __attribute__((aligned(32))) __bf16 sW[64 * 64];  // 8KB, straight copy
  {
    const uint4* wsrc = (const uint4*)Wb;
    uint4* wdst = (uint4*)sW;
#pragma unroll
    for (int idx = threadIdx.x; idx < 512; idx += 256) wdst[idx] = wsrc[idx];
  }
  __syncthreads();

  int wave = threadIdx.x >> 5;
  int lane = threadIdx.x & 31;
  int mw = wave >> 2;        // 0..1
  int nw = wave & 3;         // 0..3
  int m0 = blockIdx.x * 32 + mw * 16;
  int n0 = nw * 16;
  int r = lane & 15;
  bool hi = lane >= 16;
  int row = m0 + r;
  if (row > n - 1) row = n - 1;              // clamp loads; EXEC stays all-ones
  const __bf16* trow = T + (size_t)row * 64;
  int col = n0 + r;

  v8f acc = {};
#pragma unroll
  for (int k0 = 0; k0 < 64; k0 += 32) {
    // A 16x32: lanes 0-15 hold K {k0+0..15}; lanes 16-31 {+16..31}
    int ka = k0 + (hi ? 16 : 0);
    v8bf a0 = *(const v8bf*)(trow + ka);        // 16B global_load_b128
    v8bf a1 = *(const v8bf*)(trow + ka + 8);    // 16B global_load_b128
    v16bf a;
#pragma unroll
    for (int j = 0; j < 8; ++j) { a[j] = a0[j]; a[j + 8] = a1[j]; }
    // B 32x16: lane col, K = k0 + (hi?16:0) + 0..15 -> contiguous in transposed layout
    v16bf b = *(const v16bf*)(sW + col * 64 + k0 + (hi ? 16 : 0));  // 32B ds load
    acc = __builtin_amdgcn_wmma_f32_16x16x32_bf16(false, a, false, b,
                                                  (short)0, acc, false, false);
  }

  float bv = bias[col];
  float s1 = 0.0f, s2 = 0.0f;
  if (m0 + 16 <= n) {  // fast path: whole 16-row wave tile in range (always, N%32==0)
    float* zp = Z + ((size_t)m0 + (hi ? 8 : 0)) * 64 + col;
#pragma unroll
    for (int q = 0; q < 8; ++q) {
      float v = acc[q] + bv;
      zp[(size_t)q * 64] = v;
      s1 += v;
      s2 += v * v;
    }
  } else {
#pragma unroll
    for (int q = 0; q < 8; ++q) {
      int m = m0 + q + (hi ? 8 : 0);
      if (m < n) {
        float v = acc[q] + bv;
        Z[(size_t)m * 64 + col] = v;
        s1 += v;
        s2 += v * v;
      }
    }
  }
  atomicAdd(&sum[col], s1);
  atomicAdd(&sumsq[col], s2);
}

__global__ void k_output(const float* __restrict__ partial, const int* __restrict__ sites,
                         const float* __restrict__ w, const float* __restrict__ bp,
                         float* __restrict__ out, int n) {
  int i = blockIdx.x * blockDim.x + threadIdx.x;
  if (i >= n) return;
  out[i] = fmaf(partial[i], w[sites[i]], bp[0]);
}

extern "C" void kernel_launch(void* const* d_in, const int* in_sizes, int n_in,
                              void* d_out, int out_size, void* d_ws, size_t ws_size,
                              hipStream_t stream) {
  const float* x    = (const float*)d_in[0];
  const int*   sites= (const int*)d_in[1];
  const int*   ei   = (const int*)d_in[2];
  const float* ea   = (const float*)d_in[3];
  const float* w    = (const float*)d_in[4];
  const float* W0   = (const float*)d_in[5];
  const float* b0   = (const float*)d_in[6];
  const float* g0   = (const float*)d_in[7];
  const float* be0  = (const float*)d_in[8];
  const float* Wl   = (const float*)d_in[9];
  const float* bl   = (const float*)d_in[10];
  const float* gl   = (const float*)d_in[11];
  const float* bel  = (const float*)d_in[12];
  const float* eps  = (const float*)d_in[13];
  const float* Wp   = (const float*)d_in[14];
  const float* bp   = (const float*)d_in[15];
  float* out = (float*)d_out;

  const int n = in_sizes[0] / 10;   // 100000
  const int e = in_sizes[2] / 2;    // 1600000
  const int* src = ei;
  const int* dst = ei + e;

  // workspace carve-up (all offsets 16B aligned)
  char* p = (char*)d_ws;
  size_t nf64 = (size_t)n * 64;
  float*  H    = (float*)p;  p += nf64 * 4;
  float*  Z    = (float*)p;  p += nf64 * 4;
  float*  AGG  = (float*)p;  p += nf64 * 4;
  __bf16* Tb   = (__bf16*)p; p += nf64 * 2;
  float*  INV  = (float*)p;  p += (size_t)n * 4;
  float*  PART = (float*)p;  p += (size_t)n * 4;
  float*  SUM  = (float*)p;  p += 64 * 4;
  float*  SQ   = (float*)p;  p += 64 * 4;   // contiguous with SUM (zeroed together)
  float*  SC   = (float*)p;  p += 64 * 4;
  float*  SH   = (float*)p;  p += 64 * 4;
  __bf16* PW   = (__bf16*)p;                // 4*64*64 bf16, transposed weights

  auto blocks = [](long c) { return (unsigned)((c + TPB - 1) / TPB); };

  // prep: degrees, packed bf16 weights, accumulators
  k_zero<<<blocks(n), TPB, 0, stream>>>(INV, n);
  k_zero<<<blocks(n), TPB, 0, stream>>>(PART, n);
  k_packw<<<blocks(4 * 64 * 64), TPB, 0, stream>>>(Wl, PW, 4 * 64 * 64);
  k_degree<<<blocks(e), TPB, 0, stream>>>(dst, INV, e);
  k_invdeg<<<blocks(n), TPB, 0, stream>>>(INV, n);

  // layer 0: input linear + BN + ReLU (+ Wp segment 0 dot)
  k_zero<<<1, 128, 0, stream>>>(SUM, 128);
  k_input<<<blocks((long)n * 64), TPB, 0, stream>>>(x, sites, W0, b0, Z, SUM, SQ, n);
  k_finalize<<<1, 64, 0, stream>>>(SUM, SQ, g0, be0, SC, SH, n);
  k_bnrelu<<<blocks(n), TPB, 0, stream>>>(Z, SC, SH, H, PART, Wp, n);

  // layers 1..4
  for (int i = 0; i < 4; ++i) {
    k_zero<<<blocks((long)n * 64), TPB, 0, stream>>>(AGG, (long)n * 64);
    k_scatter<<<blocks((long)e * 16), TPB, 0, stream>>>(src, dst, ea, H, AGG, e);
    k_combine<<<blocks((long)n * 16), TPB, 0, stream>>>(H, AGG, INV, eps + i, Tb, n);
    k_zero<<<1, 128, 0, stream>>>(SUM, 128);
    k_wmma_gemm<<<(n + 31) / 32, 256, 0, stream>>>(Tb, PW + (size_t)i * 64 * 64,
                                                   bl + i * 64, Z, SUM, SQ, n);
    k_finalize<<<1, 64, 0, stream>>>(SUM, SQ, gl + i * 64, bel + i * 64, SC, SH, n);
    k_bnrelu<<<blocks(n), TPB, 0, stream>>>(Z, SC, SH, H, PART, Wp + (i + 1) * 64, n);
  }

  k_output<<<blocks(n), TPB, 0, stream>>>(PART, sites, w, bp, out, n);
}
